// JTNNDecoder_73727408603822
// MI455X (gfx1250) — matrix operand order repaired and checked by reading, not compiled, not run on hardware
//
#include <hip/hip_runtime.h>
#include <cstdint>

typedef _Float16 half_t;
typedef __attribute__((ext_vector_type(16))) _Float16 v16h;
typedef __attribute__((ext_vector_type(8)))  float    v8f;
typedef __attribute__((ext_vector_type(8)))  uint32_t v8u;
typedef __attribute__((ext_vector_type(4)))  unsigned int u32x4;
typedef __attribute__((ext_vector_type(8)))  int i32x8;
typedef __attribute__((ext_vector_type(4)))  int i32x4;

// Problem constants
constexpr int cB    = 512;
constexpr int cT    = 46;
constexpr int cTB   = 23552;      // T*B
constexpr int cTB1  = 23553;      // T*B + 1 (pad row)
constexpr int cH    = 512;
constexpr int cL    = 64;
constexpr int cV    = 1024;
constexpr int cROWS = 24064;      // T*B + B

// ---------------------------------------------------------------------------
// WMMA helpers: direct-from-global fragment loads (no LDS).
// A is (M x K) f16 row-major.  Lane l<16 holds row (l&15), K = k..k+7 and
// k+16..k+23; lane l>=16 holds same row, K offset by 8 (per CDNA5 ISA 7.12.2).
// ---------------------------------------------------------------------------
__device__ __forceinline__ v16h load_a_frag(const uint32_t* Au, int Kd, int row, int kstart) {
  const uint32_t* p = Au + (size_t)row * Kd + (kstart >> 1);
  v8u r;
  r[0] = p[0]; r[1] = p[1]; r[2] = p[2];  r[3] = p[3];
  r[4] = p[8]; r[5] = p[9]; r[6] = p[10]; r[7] = p[11];
  return __builtin_bit_cast(v16h, r);
}

// B fragments are pre-packed: frag (kt,nt) = 32 lanes x 8 dwords contiguous.
__device__ __forceinline__ v16h load_b_frag(const uint32_t* p) {
  return __builtin_bit_cast(v16h, *(const v8u*)p);
}

__device__ __forceinline__ v8f wmma_f16(v16h a, v16h b, v8f c) {
  return __builtin_amdgcn_wmma_f32_16x16x32_f16(false, a, false, b, (short)0, c,
                                                false, false);
}

// ---------------------------------------------------------------------------
// Repack f32 (K x N) weight into WMMA-B-fragment-ready f16 layout.
// Bp[((kt*(N/16)+nt)*32 + lane)*8 + i] = pack(B[k0][n], B[k0+1][n])
//   where n = nt*16 + (lane&15), koff = (lane&16)?8:0, k0 = kt*32+koff+2i+(i>=4?8:0).
// ---------------------------------------------------------------------------
__global__ __launch_bounds__(256) void repack_b(const float* __restrict__ Bsrc,
                                                uint32_t* __restrict__ Bp,
                                                int K, int N) {
  int idx = blockIdx.x * 256 + threadIdx.x;            // over K*N/2 packed dwords
  int i    = idx & 7;
  int lane = (idx >> 3) & 31;
  int f    = idx >> 8;
  int ntl  = N >> 4;
  int nt   = f % ntl;
  int kt   = f / ntl;
  int n    = nt * 16 + (lane & 15);
  int koff = (lane & 16) ? 8 : 0;
  int k0   = kt * 32 + koff + 2 * i + ((i & 4) ? 8 : 0);
  union { half_t h[2]; uint32_t u; } pk;
  pk.h[0] = (half_t)Bsrc[(size_t)k0 * N + n];
  pk.h[1] = (half_t)Bsrc[(size_t)(k0 + 1) * N + n];
  Bp[idx] = pk.u;
}

__global__ __launch_bounds__(256) void cvt16(const float* __restrict__ s,
                                             half_t* __restrict__ d, int n) {
  int i = blockIdx.x * 256 + threadIdx.x;
  if (i < n) d[i] = (half_t)s[i];
}

// ---------------------------------------------------------------------------
// Generic WMMA GEMM: C(MxN,f32) = A(MxK,f16 row-major) @ Bpacked(KxN).
// block = (32,8): 8 waves, each wave a 32x32 C tile (2x2 frags, 4 wmma/K-step).
// grid = (N/32, M/256).  mode 0: store f32.  mode 1: relu(acc+bias) -> f16.
// ---------------------------------------------------------------------------
__global__ __launch_bounds__(256) void gemm_wmma(const half_t* __restrict__ A,
                                                 const uint32_t* __restrict__ Bp,
                                                 float* __restrict__ Cf,
                                                 half_t* __restrict__ Ch,
                                                 const float* __restrict__ bias,
                                                 int M, int N, int K, int mode) {
  const int lane  = threadIdx.x;
  const int wave  = threadIdx.y;
  const int hi    = lane >> 4;
  const int l15   = lane & 15;
  const int nBase = blockIdx.x * 32;
  const int mBase = blockIdx.y * 256 + wave * 32;
  const int Kd    = K >> 1;
  const int ntl   = N >> 4;
  const uint32_t* Au = (const uint32_t*)A;

  v8f acc00 = {}, acc01 = {}, acc10 = {}, acc11 = {};
  const int kt_end = K >> 5;
  for (int kt = 0; kt < kt_end; ++kt) {
    const int k = kt << 5;
    v16h a0 = load_a_frag(Au, Kd, mBase + l15,      k + hi * 8);
    v16h a1 = load_a_frag(Au, Kd, mBase + 16 + l15, k + hi * 8);
    const uint32_t* bb = Bp + ((size_t)(kt * ntl + (nBase >> 4)) * 32 + lane) * 8;
    v16h b0 = load_b_frag(bb);
    v16h b1 = load_b_frag(bb + 256);               // next nt frag: +32 lanes * 8 dwords
    if (kt + 1 < kt_end) {
      // prefetch next K-step's B fragments into cache (global_prefetch_b8)
      __builtin_prefetch(Bp + ((size_t)((kt + 1) * ntl + (nBase >> 4)) * 32 + lane) * 8, 0, 0);
    }
    acc00 = wmma_f16(a0, b0, acc00);
    acc01 = wmma_f16(a0, b1, acc01);
    acc10 = wmma_f16(a1, b0, acc10);
    acc11 = wmma_f16(a1, b1, acc11);
  }

  // Epilogue.  D layout: VGPR j -> row = j + hi*8, col = l15 (ISA 7.12.2).
  v8f* accs[4] = {&acc00, &acc01, &acc10, &acc11};
  for (int q = 0; q < 4; ++q) {
    int r0 = mBase + (q >> 1) * 16 + hi * 8;
    int c  = nBase + (q & 1) * 16 + l15;
    v8f av = *accs[q];
    if (mode == 0) {
      for (int j = 0; j < 8; ++j) Cf[(size_t)(r0 + j) * N + c] = av[j];
    } else {
      float bv = bias[c];
      for (int j = 0; j < 8; ++j) {
        float v = av[j] + bv;
        Ch[(size_t)(r0 + j) * N + c] = (half_t)fmaxf(v, 0.f);
      }
    }
  }
}

// ---------------------------------------------------------------------------
// Scan-step kernels
// ---------------------------------------------------------------------------
__global__ __launch_bounds__(256) void step_gather1(const half_t* __restrict__ emb16,
                                                    const float* __restrict__ hbuf,
                                                    const int* __restrict__ wid,
                                                    const int* __restrict__ ids,
                                                    half_t* __restrict__ X16,
                                                    float* __restrict__ SumH,
                                                    half_t* __restrict__ A1) {
  int idx = blockIdx.x * 256 + threadIdx.x;   // B*H
  int b = idx >> 9, h = idx & 511;
  half_t x = emb16[(size_t)wid[b] * cH + h];
  X16[idx] = x;
  const int* ib = ids + b * 8;
  float s = 0.f;
  for (int n = 0; n < 8; ++n) s += hbuf[(size_t)ib[n] * cH + h];
  SumH[idx] = s;
  A1[(size_t)b * 1024 + h]       = x;
  A1[(size_t)b * 1024 + cH + h]  = (half_t)s;
}

__global__ __launch_bounds__(256) void step_gather_hn(const half_t* __restrict__ hbufH,
                                                      const int* __restrict__ ids,
                                                      half_t* __restrict__ Hn16) {
  int idx = blockIdx.x * 256 + threadIdx.x;   // B*8*H
  int bn = idx >> 9, h = idx & 511;
  Hn16[idx] = hbufH[(size_t)ids[bn] * cH + h];
}

__global__ __launch_bounds__(256) void gru_mid(const float* __restrict__ RpreX,
                                               const float* __restrict__ br,
                                               const float* __restrict__ G,
                                               const half_t* __restrict__ Hn16,
                                               const half_t* __restrict__ X16,
                                               half_t* __restrict__ A2) {
  int idx = blockIdx.x * 256 + threadIdx.x;   // B*H
  int b = idx >> 9, h = idx & 511;
  float rx = RpreX[idx] + br[h];
  float sg = 0.f;
  for (int n = 0; n < 8; ++n) {
    size_t o = ((size_t)(b * 8 + n)) * cH + h;
    float r = 1.f / (1.f + __expf(-(rx + G[o])));
    sg += r * (float)Hn16[o];
  }
  A2[(size_t)b * 1024 + h]      = X16[idx];
  A2[(size_t)b * 1024 + cH + h] = (half_t)sg;
}

__global__ __launch_bounds__(256) void gru_final(const float* __restrict__ Zpre,
                                                 const float* __restrict__ bz,
                                                 const float* __restrict__ Hpre,
                                                 const float* __restrict__ bh,
                                                 const float* __restrict__ SumH,
                                                 float* __restrict__ hbufF,
                                                 half_t* __restrict__ hbufH, int t) {
  int idx = blockIdx.x * 256 + threadIdx.x;   // B*H
  int h = idx & 511;
  float z  = 1.f / (1.f + __expf(-(Zpre[idx] + bz[h])));
  float ht = tanhf(Hpre[idx] + bh[h]);
  float nh = (1.f - z) * SumH[idx] + z * ht;
  size_t o = (size_t)t * (cB * cH) + idx;
  hbufF[o] = nh;
  hbufH[o] = (half_t)nh;
}

// ---------------------------------------------------------------------------
// Head-row construction
// ---------------------------------------------------------------------------
__global__ __launch_bounds__(256) void build_stop_rows(const int* __restrict__ wid,
                                                       const int* __restrict__ rwid,
                                                       const int* __restrict__ onei,
                                                       const int* __restrict__ ronei,
                                                       const half_t* __restrict__ emb16,
                                                       const float* __restrict__ hbuf,
                                                       const float* __restrict__ mol,
                                                       half_t* __restrict__ rows) {
  int idx = blockIdx.x * 256 + threadIdx.x;   // ROWS*1088 exactly
  int i = idx / 1088, c = idx - i * 1088;
  bool isroot = (i >= cTB);
  int b = isroot ? (i - cTB) : (i & 511);
  half_t v;
  if (c < cH) {
    int wv = isroot ? rwid[b] : wid[i];
    v = emb16[(size_t)wv * cH + c];
  } else if (c < 2 * cH) {
    const int* ib = isroot ? (ronei + b * 8) : (onei + (size_t)i * 8);
    float s = 0.f;
    for (int n = 0; n < 8; ++n) s += hbuf[(size_t)ib[n] * cH + (c - cH)];
    v = (half_t)s;
  } else {
    v = (half_t)mol[b * cL + (c - 2 * cH)];
  }
  rows[idx] = v;
}

__global__ __launch_bounds__(256) void build_pred_rows(const half_t* __restrict__ hbufH,
                                                       const float* __restrict__ mol,
                                                       half_t* __restrict__ rows) {
  int idx = blockIdx.x * 256 + threadIdx.x;   // ROWS*576 exactly
  int i = idx / 576, c = idx - i * 576;
  half_t v;
  if (c < cH) {
    v = (i < cB) ? (half_t)0.f : hbufH[(size_t)(i - cB) * cH + c];
  } else {
    int b = (i < cB) ? i : ((i - cB) & 511);
    v = (half_t)mol[b * cL + (c - cH)];
  }
  rows[idx] = v;
}

// ---------------------------------------------------------------------------
// Stop head: s = hidS16[i] . Us + bs; BCE-with-logits + accuracy.
// ---------------------------------------------------------------------------
__global__ __launch_bounds__(256) void stop_head(const half_t* __restrict__ hid,
                                                 const float* __restrict__ Us,
                                                 const float* __restrict__ bs,
                                                 const int* __restrict__ dirs,
                                                 float* __restrict__ accum) {
  int i = blockIdx.x * 256 + threadIdx.x;
  if (i >= cROWS) return;
  const half_t* row = hid + (size_t)i * cH;
  float s = bs[0];
  for (int h = 0; h < cH; ++h) s += (float)row[h] * Us[h];
  float tgt  = (i < cTB) ? (float)dirs[i] : 0.f;
  float loss = fmaxf(s, 0.f) - s * tgt + log1pf(__expf(-fabsf(s)));
  float ok   = ((s >= 0.5f) == (tgt > 0.5f)) ? 1.f : 0.f;
  atomicAdd(&accum[3], loss);
  atomicAdd(&accum[4], ok);
}

// ---------------------------------------------------------------------------
// Fused pred head: one block = 16 rows.  The 16x512 f16 A tile (16 KB,
// contiguous rows) is DMAed once into LDS by the Tensor Data Mover
// (tensor_load_to_lds, TENSORcnt), instead of 8 waves re-reading it from
// global every K-step.  WMMA vs packed Wo; logits stay in accumulators;
// softmax-CE + argmax via in-register shfl reductions (no 64KB logits LDS).
// block = (32,8), wave w owns columns [w*128, w*128+128).
// ---------------------------------------------------------------------------
__global__ __launch_bounds__(256) void pred_head(const half_t* __restrict__ Hp,
                                                 const uint32_t* __restrict__ WoP,
                                                 const float* __restrict__ bo,
                                                 const int* __restrict__ rwid,
                                                 const int* __restrict__ y_wid,
                                                 const int* __restrict__ dirs,
                                                 float* __restrict__ accum) {
  __shared__ half_t shA[16 * 512];                 // 16 KB staged A tile
  const int lane = threadIdx.x, wave = threadIdx.y;
  const int hi = lane >> 4, l15 = lane & 15;
  const int rowBase = blockIdx.x * 16;
  const int tid = wave * 32 + lane;

  if (wave == 0) {
    // Build Tensor DMA Descriptor (D#) for a 2D tile: 16 rows x 512 f16,
    // row stride 512, dense.  Groups 2/3 (and the extra group on this
    // toolchain's 6-arg builtin) unused for a 2D tensor -> zero-filled.
    unsigned long long ga = (unsigned long long)(const void*)(Hp + (size_t)rowBase * cH);
    unsigned int ldsa = (unsigned int)(unsigned long long)(void*)&shA[0];
    u32x4 g0;
    g0[0] = 1u;                                           // count=1 (valid user D#)
    g0[1] = ldsa;                                         // lds_addr
    g0[2] = (unsigned int)(ga & 0xFFFFFFFFull);           // global_addr[31:0]
    g0[3] = (unsigned int)((ga >> 32) & 0x1FFFFFFull)     // global_addr[56:32]
            | (2u << 30);                                 // type = 2 ("image")
    i32x8 g1;
    g1[0] = (1 << 16);                                    // data_size=1 -> 2 bytes
    g1[1] = (int)(512u << 16);                            // tensor_dim0[15:0]=512
    g1[2] = (int)((unsigned)cROWS << 16);                 // tensor_dim1[15:0]=24064
    g1[3] = (int)(512u << 16);                            // tile_dim0=512
    g1[4] = 16;                                           // tile_dim1=16 rows
    g1[5] = 512;                                          // tensor_dim0_stride=512
    g1[6] = 0;
    g1[7] = 0;
    i32x4 z4 = {0, 0, 0, 0};
    i32x8 z8 = {0, 0, 0, 0, 0, 0, 0, 0};
    __builtin_amdgcn_tensor_load_to_lds(g0, g1, z4, z4, z8, 0);
    __builtin_amdgcn_s_wait_tensorcnt(0);
  }
  __syncthreads();

  const uint32_t* Ash = (const uint32_t*)&shA[0];  // 256 dwords per row
  v8f acc[8] = {};
  const int ntl = 64;                              // 1024/16
  for (int kt = 0; kt < 16; ++kt) {
    const int ks = (kt << 5) + hi * 8;
    const uint32_t* p = Ash + l15 * 256 + (ks >> 1);
    v8u r;
    r[0] = p[0]; r[1] = p[1]; r[2] = p[2];  r[3] = p[3];
    r[4] = p[8]; r[5] = p[9]; r[6] = p[10]; r[7] = p[11];
    v16h a = __builtin_bit_cast(v16h, r);
    const int nt0 = wave << 3;
    for (int ni = 0; ni < 8; ++ni) {
      v16h b = load_b_frag(WoP + ((size_t)(kt * ntl + nt0 + ni) * 32 + lane) * 8);
      acc[ni] = wmma_f16(a, b, acc[ni]);
    }
  }
  for (int ni = 0; ni < 8; ++ni) {
    int col = (wave << 7) + (ni << 4) + l15;
    float bb = bo[col];
    for (int j = 0; j < 8; ++j) acc[ni][j] += bb;
  }

  __shared__ float shmax[16][8];
  __shared__ int   sharg[16][8];
  __shared__ float shsum[16][8];
  __shared__ float rowmax[16], tgtlog[16], smask[16];
  __shared__ int   rowarg[16], stgt[16];

  if (tid < 16) {
    int gi = rowBase + tid;
    if (gi < cB) { stgt[tid] = rwid[gi];       smask[tid] = 1.f; }
    else         { stgt[tid] = y_wid[gi - cB]; smask[tid] = (float)dirs[gi - cB]; }
  }
  __syncthreads();

  // per-row max / argmax / target-logit capture
  for (int j = 0; j < 8; ++j) {
    int row = j + hi * 8;
    float m = -3.0e38f; int a = 0;
    int tg = stgt[row];
    for (int ni = 0; ni < 8; ++ni) {
      float v = acc[ni][j];
      int col = (wave << 7) + (ni << 4) + l15;
      if (v > m) { m = v; a = col; }
      if (col == tg) tgtlog[row] = v;
    }
    for (int d = 1; d < 16; d <<= 1) {
      float om = __shfl_xor(m, d, 32);
      int   oa = __shfl_xor(a, d, 32);
      if (om > m) { m = om; a = oa; }
    }
    if (l15 == 0) { shmax[row][wave] = m; sharg[row][wave] = a; }
  }
  __syncthreads();
  if (tid < 16) {
    float m = shmax[tid][0]; int a = sharg[tid][0];
    for (int ww = 1; ww < 8; ++ww)
      if (shmax[tid][ww] > m) { m = shmax[tid][ww]; a = sharg[tid][ww]; }
    rowmax[tid] = m; rowarg[tid] = a;
  }
  __syncthreads();

  // per-row sum of exp(logit - max)
  for (int j = 0; j < 8; ++j) {
    int row = j + hi * 8;
    float rm = rowmax[row];
    float s = 0.f;
    for (int ni = 0; ni < 8; ++ni) s += __expf(acc[ni][j] - rm);
    for (int d = 1; d < 16; d <<= 1) s += __shfl_xor(s, d, 32);
    if (l15 == 0) shsum[row][wave] = s;
  }
  __syncthreads();
  if (tid < 16) {
    float s = 0.f;
    for (int ww = 0; ww < 8; ++ww) s += shsum[tid][ww];
    float lse = rowmax[tid] + __logf(s);
    float ce  = lse - tgtlog[tid];
    float mk  = smask[tid];
    atomicAdd(&accum[0], mk * ce);
    atomicAdd(&accum[1], mk * ((rowarg[tid] == stgt[tid]) ? 1.f : 0.f));
    atomicAdd(&accum[2], mk);
  }
}

__global__ void finalize_out(const float* __restrict__ accum, float* __restrict__ out) {
  if (threadIdx.x == 0) {
    out[0] = accum[0] / (float)cB;        // pred_loss
    out[1] = accum[3] / (float)cB;        // stop_loss
    out[2] = accum[1] / accum[2];         // pred_acc
    out[3] = accum[4] / (float)cROWS;     // stop_acc
  }
}

// ---------------------------------------------------------------------------
extern "C" void kernel_launch(void* const* d_in, const int* in_sizes, int n_in,
                              void* d_out, int out_size, void* d_ws, size_t ws_size,
                              hipStream_t stream) {
  (void)in_sizes; (void)n_in; (void)out_size; (void)ws_size;
  const float* mol   = (const float*)d_in[0];
  const int*   wid   = (const int*)d_in[1];
  const int*   y_wid = (const int*)d_in[2];
  const int*   dirs  = (const int*)d_in[3];
  const int*   hnei  = (const int*)d_in[4];
  const int*   onei  = (const int*)d_in[5];
  const int*   rwid  = (const int*)d_in[6];
  const int*   ronei = (const int*)d_in[7];
  const float* emb   = (const float*)d_in[8];
  const float* Wz    = (const float*)d_in[9];
  const float* bz    = (const float*)d_in[10];
  const float* Wr    = (const float*)d_in[11];
  const float* br    = (const float*)d_in[12];
  const float* Ur    = (const float*)d_in[13];
  const float* Wh    = (const float*)d_in[14];
  const float* bh    = (const float*)d_in[15];
  const float* Wm    = (const float*)d_in[16];
  const float* bW    = (const float*)d_in[17];
  const float* Um    = (const float*)d_in[18];
  const float* bU    = (const float*)d_in[19];
  const float* Wo    = (const float*)d_in[20];
  const float* bo    = (const float*)d_in[21];
  const float* Us    = (const float*)d_in[22];
  const float* bs    = (const float*)d_in[23];

  char* wp = (char*)d_ws;
  auto take = [&](size_t bytes) -> void* {
    void* p = (void*)wp;
    wp += (bytes + 255) & ~(size_t)255;
    return p;
  };

  float*    hbufF = (float*)   take((size_t)cTB1 * cH * 4);
  half_t*   hbufH = (half_t*)  take((size_t)cTB1 * cH * 2);
  half_t*   emb16 = (half_t*)  take((size_t)cV * cH * 2);
  uint32_t* WzP   = (uint32_t*)take((size_t)1024 * 512 * 2);
  uint32_t* WrP   = (uint32_t*)take((size_t)512 * 512 * 2);
  uint32_t* UrP   = (uint32_t*)take((size_t)512 * 512 * 2);
  uint32_t* WhP   = (uint32_t*)take((size_t)1024 * 512 * 2);
  uint32_t* UP    = (uint32_t*)take((size_t)1088 * 512 * 2);
  uint32_t* WP    = (uint32_t*)take((size_t)576 * 512 * 2);
  uint32_t* WoP   = (uint32_t*)take((size_t)512 * 1024 * 2);
  half_t*   X16   = (half_t*)  take((size_t)cB * cH * 2);
  float*    SumH  = (float*)   take((size_t)cB * cH * 4);
  half_t*   A1    = (half_t*)  take((size_t)cB * 1024 * 2);
  half_t*   A2    = (half_t*)  take((size_t)cB * 1024 * 2);
  half_t*   Hn16  = (half_t*)  take((size_t)cB * 8 * cH * 2);
  float*    Zpre  = (float*)   take((size_t)cB * cH * 4);
  float*    RpreX = (float*)   take((size_t)cB * cH * 4);
  float*    Gbuf  = (float*)   take((size_t)cB * 8 * cH * 4);
  float*    Hpre  = (float*)   take((size_t)cB * cH * 4);
  half_t*   RowsB = (half_t*)  take((size_t)cROWS * 1088 * 2);   // stop rows, then pred rows
  half_t*   HidB  = (half_t*)  take((size_t)cROWS * cH * 2);     // hidS then hidP
  float*    accum = (float*)   take(256);

  (void)hipMemsetAsync(hbufF, 0, (size_t)cTB1 * cH * 4, stream);
  (void)hipMemsetAsync(hbufH, 0, (size_t)cTB1 * cH * 2, stream);
  (void)hipMemsetAsync(accum, 0, 256, stream);

  // One-time (per call) weight conversion / WMMA packing.
  cvt16<<<(cV * cH) / 256, 256, 0, stream>>>(emb, emb16, cV * cH);
  repack_b<<<(1024 * 512 / 2) / 256, 256, 0, stream>>>(Wz, WzP, 1024, 512);
  repack_b<<<(512 * 512 / 2) / 256, 256, 0, stream>>>(Wr, WrP, 512, 512);
  repack_b<<<(512 * 512 / 2) / 256, 256, 0, stream>>>(Ur, UrP, 512, 512);
  repack_b<<<(1024 * 512 / 2) / 256, 256, 0, stream>>>(Wh, WhP, 1024, 512);
  repack_b<<<(1088 * 512 / 2) / 256, 256, 0, stream>>>(Um, UP, 1088, 512);
  repack_b<<<(576 * 512 / 2) / 256, 256, 0, stream>>>(Wm, WP, 576, 512);
  repack_b<<<(512 * 1024 / 2) / 256, 256, 0, stream>>>(Wo, WoP, 512, 1024);

  dim3 gb(32, 8);

  // Sequential GRU scan over T=46 steps.
  for (int t = 0; t < cT; ++t) {
    const int* wid_t  = wid + t * cB;
    const int* hnei_t = hnei + (size_t)t * cB * 8;
    step_gather1<<<(cB * cH) / 256, 256, 0, stream>>>(emb16, hbufF, wid_t, hnei_t,
                                                      X16, SumH, A1);
    step_gather_hn<<<(cB * 8 * cH) / 256, 256, 0, stream>>>(hbufH, hnei_t, Hn16);
    gemm_wmma<<<dim3(16, 2), gb, 0, stream>>>(A1, WzP, Zpre, nullptr, nullptr,
                                              512, 512, 1024, 0);
    gemm_wmma<<<dim3(16, 2), gb, 0, stream>>>(X16, WrP, RpreX, nullptr, nullptr,
                                              512, 512, 512, 0);
    gemm_wmma<<<dim3(16, 16), gb, 0, stream>>>(Hn16, UrP, Gbuf, nullptr, nullptr,
                                               4096, 512, 512, 0);
    gru_mid<<<(cB * cH) / 256, 256, 0, stream>>>(RpreX, br, Gbuf, Hn16, X16, A2);
    gemm_wmma<<<dim3(16, 2), gb, 0, stream>>>(A2, WhP, Hpre, nullptr, nullptr,
                                              512, 512, 1024, 0);
    gru_final<<<(cB * cH) / 256, 256, 0, stream>>>(Zpre, bz, Hpre, bh, SumH,
                                                   hbufF, hbufH, t);
  }

  // Stop head.
  build_stop_rows<<<(cROWS * 1088) / 256, 256, 0, stream>>>(wid, rwid, onei, ronei,
                                                            emb16, hbufF, mol, RowsB);
  gemm_wmma<<<dim3(16, cROWS / 256), gb, 0, stream>>>(RowsB, UP, nullptr, HidB, bU,
                                                      cROWS, 512, 1088, 1);
  stop_head<<<cROWS / 256, 256, 0, stream>>>(HidB, Us, bs, dirs, accum);

  // Pred head (reuses RowsB/HidB).
  build_pred_rows<<<(cROWS * 576) / 256, 256, 0, stream>>>(hbufH, mol, RowsB);
  gemm_wmma<<<dim3(16, cROWS / 256), gb, 0, stream>>>(RowsB, WP, nullptr, HidB, bW,
                                                      cROWS, 512, 576, 1);
  pred_head<<<cROWS / 16, gb, 0, stream>>>(HidB, WoP, bo, rwid, y_wid, dirs, accum);

  finalize_out<<<1, 32, 0, stream>>>(accum, (float*)d_out);
}
